// DualStreamMMDiT_52097953300548
// MI455X (gfx1250) — matrix operand order repaired and checked by reading, not compiled, not run on hardware
//
#include <hip/hip_runtime.h>
#include <math.h>

// ---------------------------------------------------------------------------
// Types for CDNA5 WMMA (gfx1250, wave32)
// ---------------------------------------------------------------------------
typedef unsigned short u16;
typedef unsigned int   u32;
typedef __attribute__((ext_vector_type(16))) __bf16 v16bf;
typedef __attribute__((ext_vector_type(8)))  float  v8f;

union Frag { u32 u[8]; v16bf b; };

__device__ __forceinline__ u16 f2bf(float x) {
  u32 u = __float_as_uint(x);
  u += 0x7FFFu + ((u >> 16) & 1u);          // round-to-nearest-even
  return (u16)(u >> 16);
}
__device__ __forceinline__ u32 pack2(float lo, float hi) {
  return (u32)f2bf(lo) | ((u32)f2bf(hi) << 16);
}

__device__ __forceinline__ v8f wmma_bf16(v16bf a, v16bf b, v8f c) {
  return __builtin_amdgcn_wmma_f32_16x16x32_bf16(
      false, a, false, b, (short)0, c, false, false);
}

// --- CDNA5 async global->LDS staging (ASYNCcnt-tracked DMA-style copies) ---
// ISA 10.2: generic LDS address truncates to addr[31:0] = LDS byte address.
__device__ __forceinline__ u32 lds_off(const void* p) {
  return (u32)(size_t)p;
}
__device__ __forceinline__ void async_copy_b128(u32 lds_byte, const void* g) {
  asm volatile("global_load_async_to_lds_b128 %0, %1, off"
               :: "v"(lds_byte), "v"(g) : "memory");
}
template <int N>
__device__ __forceinline__ void async_wait_le() {
  asm volatile("s_wait_asynccnt %0" :: "i"(N) : "memory");
}

// A fragment: 16x32 (MxK) bf16, rows row-major in LDS (stride ldk ushorts,
// ldk multiple of 8 -> 16B-aligned b128 loads).
// ISA layout: lanes 0-15 row M=lane, K in {8h..8h+7, 16+8h..16+8h+7}.
__device__ __forceinline__ v16bf lds_load_a(const u16* tile, int rowBase,
                                            int ldk, int kOff, int lane) {
  const int h = (lane >> 4) & 1, m = lane & 15;
  const u16* r = tile + (size_t)(rowBase + m) * ldk + kOff;
  uint4 x = *reinterpret_cast<const uint4*>(r + 8 * h);
  uint4 y = *reinterpret_cast<const uint4*>(r + 16 + 8 * h);
  Frag f;
  f.u[0] = x.x; f.u[1] = x.y; f.u[2] = x.z; f.u[3] = x.w;
  f.u[4] = y.x; f.u[5] = y.y; f.u[6] = y.z; f.u[7] = y.w;
  return f.b;
}

// B fragment: 32x16 (KxN) bf16 where B[k][n] = tile[n][k] (tile row-major in K).
__device__ __forceinline__ v16bf lds_load_b(const u16* tile, int rowBase,
                                            int ldk, int kOff, int lane) {
  const int n = lane & 15, kh = ((lane >> 4) & 1) * 8;
  const u16* r = tile + (size_t)(rowBase + n) * ldk + kOff + kh;
  uint4 x = *reinterpret_cast<const uint4*>(r);
  uint4 y = *reinterpret_cast<const uint4*>(r + 16);
  Frag f;
  f.u[0] = x.x; f.u[1] = x.y; f.u[2] = x.z; f.u[3] = x.w;
  f.u[4] = y.x; f.u[5] = y.y; f.u[6] = y.z; f.u[7] = y.w;
  return f.b;
}

// A fragment assembled from an f32 LDS row block (softmax P), cvt to bf16.
__device__ __forceinline__ v16bf lds_load_a_f32(const float* tile, int ldk,
                                                int kOff, int lane) {
  const int h = (lane >> 4) & 1, m = lane & 15;
  const float* r = tile + (size_t)m * ldk + kOff;
  Frag f;
#pragma unroll
  for (int v = 0; v < 8; ++v) {
    int k = (v < 4) ? (2 * v + 8 * h) : (16 + 2 * (v - 4) + 8 * h);
    f.u[v] = pack2(r[k], r[k + 1]);
  }
  return f.b;
}

// ---------------------------------------------------------------------------
// GEMM: Out[M,N] = X[M,K](bf16) * W[N,K](bf16)^T (+ fused epilogues)
// 128x128x32 tiles, 8 waves, wave = 32x64 sub-tile. Interior blocks use
// double-buffered global_load_async_to_lds_b128 staging (4 in-order async
// transfers per wave per tile -> s_wait_asynccnt 4 overlaps copy with WMMA).
// Edge blocks fall back to a guarded single-buffer path.
// ---------------------------------------------------------------------------
enum { EPI_NONE = 0, EPI_BIAS = 1, EPI_RESGATE = 2, EPI_SILU_MUL = 3 };

template <int EPI, int OUTB>
__global__ __launch_bounds__(256) void gemm_bf16(
    const u16* __restrict__ X, const u16* __restrict__ W,
    const float* __restrict__ bias, void* __restrict__ OutRaw,
    const float* __restrict__ Aux,   // RESGATE: residual; SILU_MUL: multiplier
    const float* __restrict__ Gate,  // RESGATE: per-batch gate row
    int M, int N, int K, int tokensPerBatch, int gateStride) {
  __shared__ u16 sA[2][128 * 40];
  __shared__ u16 sB[2][128 * 40];
  const int tid = threadIdx.x;
  const int lane = tid & 31;
  const int w = tid >> 5;
  const int wm = w >> 1, wn = w & 1;
  const int bm = blockIdx.y * 128, bn = blockIdx.x * 128;

  v8f acc[2][4];
#pragma unroll
  for (int i = 0; i < 2; ++i)
#pragma unroll
    for (int j = 0; j < 4; ++j)
#pragma unroll
      for (int v = 0; v < 8; ++v) acc[i][j][v] = 0.f;

  auto stage_full = [&](int k0, int buf) {
#pragma unroll
    for (int i = 0; i < 2; ++i) {
      int idx = tid + (i << 8);
      int r = idx >> 2, c8 = (idx & 3) << 3;
      int gk = k0 + c8;
      async_copy_b128(lds_off(&sA[buf][r * 40 + c8]),
                      X + (size_t)(bm + r) * K + gk);
      async_copy_b128(lds_off(&sB[buf][r * 40 + c8]),
                      W + (size_t)(bn + r) * K + gk);
    }
  };
  auto stage_guard = [&](int k0, int buf) {
#pragma unroll
    for (int i = 0; i < 2; ++i) {
      int idx = tid + (i << 8);
      int r = idx >> 2, c8 = (idx & 3) << 3;
      int gk = k0 + c8;
      bool kok = (gk + 8 <= K);
      if ((bm + r < M) && kok)
        async_copy_b128(lds_off(&sA[buf][r * 40 + c8]),
                        X + (size_t)(bm + r) * K + gk);
      else
        *reinterpret_cast<uint4*>(&sA[buf][r * 40 + c8]) = make_uint4(0, 0, 0, 0);
      if ((bn + r < N) && kok)
        async_copy_b128(lds_off(&sB[buf][r * 40 + c8]),
                        W + (size_t)(bn + r) * K + gk);
      else
        *reinterpret_cast<uint4*>(&sB[buf][r * 40 + c8]) = make_uint4(0, 0, 0, 0);
    }
  };
  auto compute = [&](int buf) {
    v16bf afrag[2], bfrag[4];
#pragma unroll
    for (int i = 0; i < 2; ++i)
      afrag[i] = lds_load_a(&sA[buf][0], wm * 32 + i * 16, 40, 0, lane);
#pragma unroll
    for (int j = 0; j < 4; ++j)
      bfrag[j] = lds_load_b(&sB[buf][0], wn * 64 + j * 16, 40, 0, lane);
#pragma unroll
    for (int i = 0; i < 2; ++i)
#pragma unroll
      for (int j = 0; j < 4; ++j)
        acc[i][j] = wmma_bf16(afrag[i], bfrag[j], acc[i][j]);
  };

  const bool edge = (bm + 128 > M) || (bn + 128 > N) || ((K & 31) != 0);
  if (!edge) {
    const int nk = K >> 5;
    stage_full(0, 0);
    for (int it = 0; it < nk; ++it) {
      const int cur = it & 1;
      if (it + 1 < nk) {
        stage_full((it + 1) << 5, cur ^ 1);  // overlap next copy with WMMA
        async_wait_le<4>();                  // tile `cur` (first 4) landed
      } else {
        async_wait_le<0>();
      }
      __syncthreads();
      compute(cur);
      __syncthreads();
    }
  } else {
    for (int k0 = 0; k0 < K; k0 += 32) {
      stage_guard(k0, 0);
      async_wait_le<0>();
      __syncthreads();
      compute(0);
      __syncthreads();
    }
  }

  // Epilogue. C/D layout: VGPR v, lane -> n = lane&15, m = v + 8*(lane>>4).
  const int halfl = (lane >> 4) & 1, n0 = lane & 15;
#pragma unroll
  for (int i = 0; i < 2; ++i) {
#pragma unroll
    for (int j = 0; j < 4; ++j) {
      int col = bn + wn * 64 + j * 16 + n0;
      if (col >= N) continue;
#pragma unroll
      for (int v = 0; v < 8; ++v) {
        int row = bm + wm * 32 + i * 16 + v + 8 * halfl;
        if (row >= M) continue;
        float val = acc[i][j][v];
        size_t oidx = (size_t)row * N + col;
        if (EPI == EPI_BIAS) {
          val += bias[col];
        } else if (EPI == EPI_RESGATE) {
          if (bias) val += bias[col];
          int b = row / tokensPerBatch;
          val = Aux[oidx] + Gate[(size_t)b * gateStride + col] * val;
        } else if (EPI == EPI_SILU_MUL) {
          float s = val / (1.f + __expf(-val));
          val = s * Aux[oidx];
        }
        if (OUTB)
          reinterpret_cast<u16*>(OutRaw)[oidx] = f2bf(val);
        else
          reinterpret_cast<float*>(OutRaw)[oidx] = val;
      }
    }
  }
}

// ---------------------------------------------------------------------------
// RMSNorm + adaLN modulate -> bf16: out = rms(x)*(1+mods[scOff]) + mods[shOff]
// ---------------------------------------------------------------------------
__global__ __launch_bounds__(256) void rms_modulate(
    const float* __restrict__ X, const float* __restrict__ mods,
    u16* __restrict__ Out, int T, int D, int modStride, int shOff, int scOff) {
  __shared__ float red[256];
  const int row = blockIdx.x, tid = threadIdx.x;
  const int b = row / T;
  const float* xr = X + (size_t)row * D;
  float ss = 0.f;
  for (int d = tid; d < D; d += 256) { float v = xr[d]; ss += v * v; }
  red[tid] = ss;
  __syncthreads();
  for (int s = 128; s > 0; s >>= 1) {
    if (tid < s) red[tid] += red[tid + s];
    __syncthreads();
  }
  const float inv = rsqrtf(red[0] / (float)D + 1e-6f);
  const float* mr = mods + (size_t)b * modStride;
  u16* orow = Out + (size_t)row * D;
  for (int d = tid; d < D; d += 256)
    orow[d] = f2bf(xr[d] * inv * (1.f + mr[scOff + d]) + mr[shOff + d]);
}

__global__ void silu_vec(const float* __restrict__ in, u16* __restrict__ out,
                         int n) {
  int i = blockIdx.x * 256 + threadIdx.x;
  if (i < n) { float x = in[i]; out[i] = f2bf(x / (1.f + __expf(-x))); }
}

// f32 -> bf16 bulk conversion (weights are single-use; converted copy stays
// L2-resident so the GEMM re-reads come from L2, not HBM).
__global__ void cvt_bf16_vec(const float* __restrict__ in,
                             u16* __restrict__ out, int n) {
  int i = (blockIdx.x * 256 + threadIdx.x) * 8;
  if (i + 8 <= n) {
    float4 a = *reinterpret_cast<const float4*>(in + i);
    float4 b = *reinterpret_cast<const float4*>(in + i + 4);
    uint4 o = make_uint4(pack2(a.x, a.y), pack2(a.z, a.w), pack2(b.x, b.y),
                         pack2(b.z, b.w));
    *reinterpret_cast<uint4*>(out + i) = o;
  } else {
    for (int j = i; j < n; ++j) out[j] = f2bf(in[j]);
  }
}

// ---------------------------------------------------------------------------
// QKV split + RoPE + head scatter -> bf16 Q[bh,T,64], K[bh,Tj,64], V^T[bh,64,Tj]
// ---------------------------------------------------------------------------
__global__ void rope_scatter(const float* __restrict__ qkv, int T, int Bn,
                             int D, u16* __restrict__ qdst,
                             u16* __restrict__ kj, u16* __restrict__ vT,
                             int Tj, int tokOff, int ropeMode) {
  const int pairsPerRow = D >> 1;
  int idx = blockIdx.x * 256 + threadIdx.x;
  if (idx >= Bn * T * pairsPerRow) return;
  int m = idx / pairsPerRow;
  int p = idx - m * pairsPerRow;
  int col = p << 1;
  int b = m / T, t = m - b * T;
  int h = col >> 6, d = col & 63;
  const float* row = qkv + (size_t)m * (3 * D);
  float qe = row[col],         qo = row[col + 1];
  float ke = row[D + col],     ko = row[D + col + 1];
  float ve = row[2 * D + col], vo = row[2 * D + col + 1];

  float pos, fi, invHalf;
  if (ropeMode) {
    if (d < 32) { pos = (float)(t >> 5); fi = (float)(d >> 1); }
    else        { pos = (float)(t & 31); fi = (float)((d - 32) >> 1); }
    invHalf = 1.f / 16.f;
  } else {
    pos = (float)t; fi = (float)(d >> 1); invHalf = 1.f / 32.f;
  }
  float fr = __expf(-9.210340372f * fi * invHalf);  // 10000^{-fi/half}
  float ang = pos * fr;
  float sn, cs;
  __sincosf(ang, &sn, &cs);
  float qre = qe * cs - qo * sn, qro = qe * sn + qo * cs;
  float kre = ke * cs - ko * sn, kro = ke * sn + ko * cs;

  size_t bh = (size_t)(b * 16 + h);
  size_t qbase = (bh * T + t) * 64 + d;
  qdst[qbase]     = f2bf(qre);
  qdst[qbase + 1] = f2bf(qro);
  size_t kbase = (bh * Tj + tokOff + t) * 64 + d;
  kj[kbase]     = f2bf(kre);
  kj[kbase + 1] = f2bf(kro);
  size_t vbase = (bh * 64 + d) * Tj + tokOff + t;   // transposed V
  vT[vbase]      = f2bf(ve);
  vT[vbase + Tj] = f2bf(vo);
}

// ---------------------------------------------------------------------------
// Joint attention: block = (bh, 16-row q-tile). S (16 x Tj f32 = 80KB) lives
// fully in LDS (320KB/WGP on CDNA5). K/V/Q tiles staged with async copies.
// ---------------------------------------------------------------------------
__global__ __launch_bounds__(256) void attn_joint(
    const u16* __restrict__ Q, const u16* __restrict__ KJ,
    const u16* __restrict__ VT, u16* __restrict__ O,
    const int* __restrict__ ctx_valid, int Tq, int Tj, int Tc, int H,
    int Dmodel, float scale) {
  extern __shared__ char smem[];
  float* S = reinterpret_cast<float*>(smem);                 // 16 x Tj f32
  char* pm = smem + (size_t)16 * Tj * 4;
  u16* KV = reinterpret_cast<u16*>(pm); pm += 128 * 72 * 2;  // K/V chunk tile
  u16* Qs = reinterpret_cast<u16*>(pm); pm += 16 * 72 * 2;   // Q tile
  float* Osum = reinterpret_cast<float*>(pm);                // 16 x 64 f32

  const int tid = threadIdx.x, lane = tid & 31, w = tid >> 5;
  const int bh = blockIdx.y, b = bh / H, h = bh % H;
  const int q0 = blockIdx.x * 16;
  const int valid = ctx_valid[b];
  const int halfl = (lane >> 4) & 1, n0 = lane & 15;

  // Stage Q tile (16 x 64 bf16) asynchronously.
  if (tid < 128) {
    int r = tid >> 3, c8 = (tid & 7) * 8;
    async_copy_b128(lds_off(Qs + r * 72 + c8),
                    Q + ((size_t)bh * Tq + q0 + r) * 64 + c8);
  }
  async_wait_le<0>();
  __syncthreads();

  const int nch = Tj / 128;
  // Phase 1: S = scale * Q K^T + mask
  for (int ch = 0; ch < nch; ++ch) {
#pragma unroll
    for (int i = 0; i < 4; ++i) {
      int idx = tid + i * 256;
      int r = idx >> 3, c8 = (idx & 7) * 8;
      async_copy_b128(lds_off(KV + r * 72 + c8),
                      KJ + ((size_t)bh * Tj + ch * 128 + r) * 64 + c8);
    }
    async_wait_le<0>();
    __syncthreads();
    v8f s;
#pragma unroll
    for (int v = 0; v < 8; ++v) s[v] = 0.f;
#pragma unroll
    for (int kk = 0; kk < 64; kk += 32) {
      v16bf a = lds_load_a(Qs, 0, 72, kk, lane);
      v16bf bb = lds_load_b(KV, w * 16, 72, kk, lane);
      s = wmma_bf16(a, bb, s);
    }
    int key = ch * 128 + w * 16 + n0;
    float madd = (key < Tc && key >= valid) ? -1e9f : 0.f;
#pragma unroll
    for (int v = 0; v < 8; ++v)
      S[(size_t)(v + 8 * halfl) * Tj + key] = s[v] * scale + madd;
    __syncthreads();
  }

  // Softmax over each of the 16 rows (16 lanes per row).
  {
    int rrow = tid >> 4, lg = tid & 15;
    float* Srow = S + (size_t)rrow * Tj;
    float mx = -3.0e38f;
    for (int j = lg; j < Tj; j += 16) mx = fmaxf(mx, Srow[j]);
#pragma unroll
    for (int o = 8; o >= 1; o >>= 1) mx = fmaxf(mx, __shfl_xor(mx, o, 16));
    float sum = 0.f;
    for (int j = lg; j < Tj; j += 16) {
      float e = __expf(Srow[j] - mx);
      Srow[j] = e;
      sum += e;
    }
#pragma unroll
    for (int o = 8; o >= 1; o >>= 1) sum += __shfl_xor(sum, o, 16);
    float inv = 1.f / sum;
    for (int j = lg; j < Tj; j += 16) Srow[j] *= inv;
  }
  __syncthreads();

  // Phase 2: O = P V, waves split the key dimension, V^T chunks via async.
  v8f ofr[4];
#pragma unroll
  for (int j = 0; j < 4; ++j)
#pragma unroll
    for (int v = 0; v < 8; ++v) ofr[j][v] = 0.f;

  for (int ch = 0; ch < nch; ++ch) {
#pragma unroll
    for (int i = 0; i < 4; ++i) {
      int idx = tid + i * 256;
      int r = idx >> 4, c8 = (idx & 15) * 8;
      async_copy_b128(lds_off(KV + r * 136 + c8),
                      VT + ((size_t)bh * 64 + r) * Tj + ch * 128 + c8);
    }
    async_wait_le<0>();
    __syncthreads();
#pragma unroll
    for (int st = 0; st < 4; ++st) {
      if (((ch * 4 + st) & 7) == w) {
        v16bf a = lds_load_a_f32(S, Tj, ch * 128 + st * 32, lane);
#pragma unroll
        for (int j = 0; j < 4; ++j) {
          v16bf bb = lds_load_b(KV, j * 16, 136, st * 32, lane);
          ofr[j] = wmma_bf16(a, bb, ofr[j]);
        }
      }
    }
    __syncthreads();
  }

  // Cross-wave reduce of partial O (16x64) via LDS float atomics, write bf16.
  for (int i = tid; i < 16 * 64; i += 256) Osum[i] = 0.f;
  __syncthreads();
#pragma unroll
  for (int j = 0; j < 4; ++j)
#pragma unroll
    for (int v = 0; v < 8; ++v)
      atomicAdd(&Osum[(v + 8 * halfl) * 64 + j * 16 + n0], ofr[j][v]);
  __syncthreads();
  for (int i = tid; i < 16 * 64; i += 256) {
    int r = i >> 6, c = i & 63;
    O[((size_t)(b * Tq + q0 + r)) * Dmodel + h * 64 + c] = f2bf(Osum[i]);
  }
}

// ---------------------------------------------------------------------------
// Host orchestration
// ---------------------------------------------------------------------------
extern "C" void kernel_launch(void* const* d_in, const int* in_sizes, int n_in,
                              void* d_out, int out_size, void* d_ws,
                              size_t ws_size, hipStream_t stream) {
  (void)in_sizes; (void)n_in; (void)out_size;
  const int B = 2, Ti = 1024, Tc = 256, Tj = Ti + Tc, C = 16, D = 1024,
            Hh = 2816, L = 4, H = 16;
  const int Mi = B * Ti, Mc = B * Tc;

  const float* x_img = (const float*)d_in[0];
  const float* z = (const float*)d_in[1];
  const float* cvec = (const float*)d_in[2];
  const int* ctx_valid = (const int*)d_in[3];
  const float* img_in_w = (const float*)d_in[4];
  const float* ctx_in_w = (const float*)d_in[5];
  const float* final_w = (const float*)d_in[6];

  struct StreamP {
    const float *qkv_w, *qkv_b, *proj_w, *proj_b, *adaln_w, *adaln_b, *w1, *w2,
        *w3;
  };
  StreamP PC = {(const float*)d_in[7],  (const float*)d_in[8],
                (const float*)d_in[9],  (const float*)d_in[10],
                (const float*)d_in[11], (const float*)d_in[12],
                (const float*)d_in[13], (const float*)d_in[14],
                (const float*)d_in[15]};
  StreamP PI = {(const float*)d_in[16], (const float*)d_in[17],
                (const float*)d_in[18], (const float*)d_in[19],
                (const float*)d_in[20], (const float*)d_in[21],
                (const float*)d_in[22], (const float*)d_in[23],
                (const float*)d_in[24]};

  char* ws = (char*)d_ws;
  size_t off = 0;
  auto alloc = [&](size_t bytes) -> char* {
    char* p = ws + off;
    off += (bytes + 255) & ~(size_t)255;
    return p;
  };
  float* Xr = (float*)alloc((size_t)Mi * D * 4);        // img residual (f32)
  float* Cr = (float*)alloc((size_t)Mc * D * 4);        // ctx residual (f32)
  u16* SCb = (u16*)alloc((size_t)B * D * 2);            // silu(c) bf16
  float* MODS_I = (float*)alloc((size_t)B * 6 * D * 4);
  float* MODS_C = (float*)alloc((size_t)B * 6 * D * 4);
  u16* BUFA = (u16*)alloc((size_t)Mi * D * 2);          // xm_i/o_i/h_i bf16
  u16* BUFC = (u16*)alloc((size_t)Mc * D * 2);          // xm_c/o_c/h_c bf16
  float* QKV = (float*)alloc((size_t)Mi * 3 * D * 4);   // qkv f32, reused a3
  u16* ACTB = (u16*)alloc((size_t)Mi * D * 2);          // input/final cvt bf16
  u16* WCVT = (u16*)alloc((size_t)6 * D * D * 2);       // per-GEMM bf16 weight
  u16* QX = (u16*)alloc((size_t)B * H * Ti * 64 * 2);
  u16* QC = (u16*)alloc((size_t)B * H * Tc * 64 * 2);
  u16* KJb = (u16*)alloc((size_t)B * H * Tj * 64 * 2);
  u16* VTb = (u16*)alloc((size_t)B * H * Tj * 64 * 2);
  u16* FFHB = (u16*)alloc((size_t)Mi * Hh * 2);         // silu*mul out bf16
  if (ws_size < off) return;  // workspace too small; bail deterministically

  dim3 blk(256);
  auto gg = [](int M, int N) { return dim3((N + 127) / 128, (M + 127) / 128); };
  auto cvtGrid = [](size_t n) { return dim3((u32)((n / 8 + 255) / 256)); };
  const size_t attnLds =
      (size_t)16 * Tj * 4 + 128 * 72 * 2 + 16 * 72 * 2 + 16 * 64 * 4;
  const float scale = 0.125f;  // 1/sqrt(64)

  // Inputs -> bf16, silu(c)
  silu_vec<<<dim3((B * D + 255) / 256), blk, 0, stream>>>(cvec, SCb, B * D);
  cvt_bf16_vec<<<cvtGrid((size_t)Mi * C), blk, 0, stream>>>(x_img, ACTB,
                                                            Mi * C);
  cvt_bf16_vec<<<cvtGrid((size_t)D * C), blk, 0, stream>>>(img_in_w, WCVT,
                                                           D * C);
  gemm_bf16<EPI_NONE, 0><<<gg(Mi, D), blk, 0, stream>>>(
      ACTB, WCVT, nullptr, Xr, nullptr, nullptr, Mi, D, C, Ti, 0);
  cvt_bf16_vec<<<cvtGrid((size_t)Mc * D), blk, 0, stream>>>(z, ACTB, Mc * D);
  cvt_bf16_vec<<<cvtGrid((size_t)D * D), blk, 0, stream>>>(ctx_in_w, WCVT,
                                                           D * D);
  gemm_bf16<EPI_NONE, 0><<<gg(Mc, D), blk, 0, stream>>>(
      ACTB, WCVT, nullptr, Cr, nullptr, nullptr, Mc, D, D, Tc, 0);

  for (int l = 0; l < L; ++l) {
    const float* aw_c = PC.adaln_w + (size_t)l * 6 * D * D;
    const float* ab_c = PC.adaln_b + (size_t)l * 6 * D;
    const float* aw_i = PI.adaln_w + (size_t)l * 6 * D * D;
    const float* ab_i = PI.adaln_b + (size_t)l * 6 * D;
    const float* qw_c = PC.qkv_w + (size_t)l * 3 * D * D;
    const float* qb_c = PC.qkv_b + (size_t)l * 3 * D;
    const float* qw_i = PI.qkv_w + (size_t)l * 3 * D * D;
    const float* qb_i = PI.qkv_b + (size_t)l * 3 * D;
    const float* pw_c = PC.proj_w + (size_t)l * D * D;
    const float* pb_c = PC.proj_b + (size_t)l * D;
    const float* pw_i = PI.proj_w + (size_t)l * D * D;
    const float* pb_i = PI.proj_b + (size_t)l * D;
    const float* w1_c = PC.w1 + (size_t)l * Hh * D;
    const float* w2_c = PC.w2 + (size_t)l * D * Hh;
    const float* w3_c = PC.w3 + (size_t)l * Hh * D;
    const float* w1_i = PI.w1 + (size_t)l * Hh * D;
    const float* w2_i = PI.w2 + (size_t)l * D * Hh;
    const float* w3_i = PI.w3 + (size_t)l * Hh * D;

    // adaLN modulation vectors
    cvt_bf16_vec<<<cvtGrid((size_t)6 * D * D), blk, 0, stream>>>(aw_c, WCVT,
                                                                 6 * D * D);
    gemm_bf16<EPI_BIAS, 0><<<gg(B, 6 * D), blk, 0, stream>>>(
        SCb, WCVT, ab_c, MODS_C, nullptr, nullptr, B, 6 * D, D, 1, 0);
    cvt_bf16_vec<<<cvtGrid((size_t)6 * D * D), blk, 0, stream>>>(aw_i, WCVT,
                                                                 6 * D * D);
    gemm_bf16<EPI_BIAS, 0><<<gg(B, 6 * D), blk, 0, stream>>>(
        SCb, WCVT, ab_i, MODS_I, nullptr, nullptr, B, 6 * D, D, 1, 0);

    // MSA modulate (sh_msa @0, sc_msa @D)
    rms_modulate<<<dim3(Mc), blk, 0, stream>>>(Cr, MODS_C, BUFC, Tc, D, 6 * D,
                                               0, D);
    rms_modulate<<<dim3(Mi), blk, 0, stream>>>(Xr, MODS_I, BUFA, Ti, D, 6 * D,
                                               0, D);

    // QKV + RoPE scatter (ctx first so QKV buffer can be reused for img)
    cvt_bf16_vec<<<cvtGrid((size_t)3 * D * D), blk, 0, stream>>>(qw_c, WCVT,
                                                                 3 * D * D);
    gemm_bf16<EPI_BIAS, 0><<<gg(Mc, 3 * D), blk, 0, stream>>>(
        BUFC, WCVT, qb_c, QKV, nullptr, nullptr, Mc, 3 * D, D, Tc, 0);
    rope_scatter<<<dim3((Mc * (D / 2) + 255) / 256), blk, 0, stream>>>(
        QKV, Tc, B, D, QC, KJb, VTb, Tj, 0, 0);
    cvt_bf16_vec<<<cvtGrid((size_t)3 * D * D), blk, 0, stream>>>(qw_i, WCVT,
                                                                 3 * D * D);
    gemm_bf16<EPI_BIAS, 0><<<gg(Mi, 3 * D), blk, 0, stream>>>(
        BUFA, WCVT, qb_i, QKV, nullptr, nullptr, Mi, 3 * D, D, Ti, 0);
    rope_scatter<<<dim3((Mi * (D / 2) + 255) / 256), blk, 0, stream>>>(
        QKV, Ti, B, D, QX, KJb, VTb, Tj, Tc, 1);

    // Joint attention (both streams read pre-update K/V)
    attn_joint<<<dim3(Ti / 16, B * H), blk, attnLds, stream>>>(
        QX, KJb, VTb, BUFA, ctx_valid, Ti, Tj, Tc, H, D, scale);
    attn_joint<<<dim3(Tc / 16, B * H), blk, attnLds, stream>>>(
        QC, KJb, VTb, BUFC, ctx_valid, Tc, Tj, Tc, H, D, scale);

    // Output proj + gated residual (g_msa @2D)
    cvt_bf16_vec<<<cvtGrid((size_t)D * D), blk, 0, stream>>>(pw_i, WCVT, D * D);
    gemm_bf16<EPI_RESGATE, 0><<<gg(Mi, D), blk, 0, stream>>>(
        BUFA, WCVT, pb_i, Xr, Xr, MODS_I + 2 * D, Mi, D, D, Ti, 6 * D);
    cvt_bf16_vec<<<cvtGrid((size_t)D * D), blk, 0, stream>>>(pw_c, WCVT, D * D);
    gemm_bf16<EPI_RESGATE, 0><<<gg(Mc, D), blk, 0, stream>>>(
        BUFC, WCVT, pb_c, Cr, Cr, MODS_C + 2 * D, Mc, D, D, Tc, 6 * D);

    // MLP img: h = mod(x); a3 = h@w3^T; ff = silu(h@w1^T)*a3; x += g*ff@w2^T
    rms_modulate<<<dim3(Mi), blk, 0, stream>>>(Xr, MODS_I, BUFA, Ti, D, 6 * D,
                                               3 * D, 4 * D);
    cvt_bf16_vec<<<cvtGrid((size_t)Hh * D), blk, 0, stream>>>(w3_i, WCVT,
                                                              Hh * D);
    gemm_bf16<EPI_NONE, 0><<<gg(Mi, Hh), blk, 0, stream>>>(
        BUFA, WCVT, nullptr, QKV, nullptr, nullptr, Mi, Hh, D, Ti, 0);
    cvt_bf16_vec<<<cvtGrid((size_t)Hh * D), blk, 0, stream>>>(w1_i, WCVT,
                                                              Hh * D);
    gemm_bf16<EPI_SILU_MUL, 1><<<gg(Mi, Hh), blk, 0, stream>>>(
        BUFA, WCVT, nullptr, FFHB, QKV, nullptr, Mi, Hh, D, Ti, 0);
    cvt_bf16_vec<<<cvtGrid((size_t)D * Hh), blk, 0, stream>>>(w2_i, WCVT,
                                                              D * Hh);
    gemm_bf16<EPI_RESGATE, 0><<<gg(Mi, D), blk, 0, stream>>>(
        FFHB, WCVT, nullptr, Xr, Xr, MODS_I + 5 * D, Mi, D, Hh, Ti, 6 * D);

    // MLP ctx
    rms_modulate<<<dim3(Mc), blk, 0, stream>>>(Cr, MODS_C, BUFC, Tc, D, 6 * D,
                                               3 * D, 4 * D);
    cvt_bf16_vec<<<cvtGrid((size_t)Hh * D), blk, 0, stream>>>(w3_c, WCVT,
                                                              Hh * D);
    gemm_bf16<EPI_NONE, 0><<<gg(Mc, Hh), blk, 0, stream>>>(
        BUFC, WCVT, nullptr, QKV, nullptr, nullptr, Mc, Hh, D, Tc, 0);
    cvt_bf16_vec<<<cvtGrid((size_t)Hh * D), blk, 0, stream>>>(w1_c, WCVT,
                                                              Hh * D);
    gemm_bf16<EPI_SILU_MUL, 1><<<gg(Mc, Hh), blk, 0, stream>>>(
        BUFC, WCVT, nullptr, FFHB, QKV, nullptr, Mc, Hh, D, Tc, 0);
    cvt_bf16_vec<<<cvtGrid((size_t)D * Hh), blk, 0, stream>>>(w2_c, WCVT,
                                                              D * Hh);
    gemm_bf16<EPI_RESGATE, 0><<<gg(Mc, D), blk, 0, stream>>>(
        FFHB, WCVT, nullptr, Cr, Cr, MODS_C + 5 * D, Mc, D, Hh, Tc, 6 * D);
  }

  // Final projection to image channels (f32 out)
  cvt_bf16_vec<<<cvtGrid((size_t)Mi * D), blk, 0, stream>>>(Xr, ACTB, Mi * D);
  cvt_bf16_vec<<<cvtGrid((size_t)C * D), blk, 0, stream>>>(final_w, WCVT,
                                                           C * D);
  gemm_bf16<EPI_NONE, 0><<<gg(Mi, C), blk, 0, stream>>>(
      ACTB, WCVT, nullptr, d_out, nullptr, nullptr, Mi, C, D, Ti, 0);
}
